// CausalSelfAttention_70789650973109
// MI455X (gfx1250) — matrix operand order repaired and checked by reading, not compiled, not run on hardware
//
#include <hip/hip_runtime.h>
#include <hip/hip_bf16.h>

#define B_    4
#define T_    2048
#define DIM_  1024
#define H_    8
#define D_    128
#define HDIM_ 1024

typedef __bf16 bf16_t;
typedef __attribute__((ext_vector_type(16))) __bf16 v16bf;
typedef __attribute__((ext_vector_type(8)))  __bf16 v8bf;
typedef __attribute__((ext_vector_type(8)))  float  v8f;

static __device__ __forceinline__ v8f wmma_bf16(v16bf a, v16bf b, v8f c) {
  // D = A(16x32) * B(32x16) + C, f32 accumulate — v_wmma_f32_16x16x32_bf16
  return __builtin_amdgcn_wmma_f32_16x16x32_bf16(false, a, false, b, (short)0, c, false, false);
}

// A-matrix 16x32 bf16 fragment (same k order as load_bfrag): lane = row (lane&15),
// lanes 0-15 hold K 0..15, lanes 16-31 hold K 16..31 (contiguous per lane).
static __device__ __forceinline__ v16bf load_afrag(const bf16_t* base, long ld, int lane) {
  int row = lane & 15;
  int c0  = (lane & 16) ? 16 : 0;
  const bf16_t* p = base + (long)row * ld;
  v8bf lo = *(const v8bf*)(p + c0);
  v8bf hi = *(const v8bf*)(p + 8 + c0);
  v16bf a;
#pragma unroll
  for (int i = 0; i < 8; ++i) { a[i] = lo[i]; a[8 + i] = hi[i]; }
  return a;
}

// B-matrix 32x16 bf16 fragment: lane = column (lane&15),
// lanes 0-15 hold K 0..15, lanes 16-31 hold K 16..31 (contiguous per lane).
// base points at Bmat[k=0][n=0] where Bmat[kk][n] = base[n*ld + kk].
static __device__ __forceinline__ v16bf load_bfrag(const bf16_t* base, long ld, int lane) {
  int col = lane & 15;
  int kb  = (lane & 16) ? 16 : 0;
  return *(const v16bf*)(base + (long)col * ld + kb);
}

// ---------------------------------------------------------------- convert
__global__ void cvt_f32_to_bf16(const float* __restrict__ src, bf16_t* __restrict__ dst, long n) {
  long i = (long)blockIdx.x * blockDim.x + threadIdx.x;
  if (i < n) dst[i] = (bf16_t)src[i];
}

// ---------------------------------------------------------------- GEMM: C = A * W^T
// A: M x K bf16 row-major; W: N x K bf16 row-major; C: M x N f32.
// One wave computes a 32x64 C macro-tile: 2 A-frags x 4 B-frags -> 8 WMMAs
// per 32-deep k-step. All 6 fragment loads of a k-step are issued before the
// 8 WMMAs so they clause together and overlap with the matrix pipe.
__global__ void gemm_bf16_abt(const bf16_t* __restrict__ A, const bf16_t* __restrict__ W,
                              float* __restrict__ C, int M, int N, int K) {
  int lane = threadIdx.x & 31;
  int wave = threadIdx.x >> 5;
  int tiles_n = N >> 6;                       // 64-wide N tiles
  long tile = (long)blockIdx.x * 4 + wave;
  int nt = (int)(tile % tiles_n);
  int mt = (int)(tile / tiles_n);
  if (mt >= (M >> 5)) return;                 // 32-tall M tiles

  const bf16_t* A0 = A + (long)(mt * 32) * K;
  const bf16_t* A1 = A0 + (long)16 * K;
  const bf16_t* W0 = W + (long)(nt * 64) * K;

  v8f acc[2][4];
#pragma unroll
  for (int mi = 0; mi < 2; ++mi)
#pragma unroll
    for (int j = 0; j < 4; ++j) acc[mi][j] = (v8f){};

  for (int k0 = 0; k0 < K; k0 += 32) {
    __builtin_prefetch(A0 + k0 + 128, 0, 0);   // global_prefetch_b8
    // issue all fragment loads up front (load clause), then the WMMA burst
    v16bf a0 = load_afrag(A0 + k0, K, lane);
    v16bf a1 = load_afrag(A1 + k0, K, lane);
    v16bf b0 = load_bfrag(W0 + (long)(0 * 16) * K + k0, K, lane);
    v16bf b1 = load_bfrag(W0 + (long)(1 * 16) * K + k0, K, lane);
    v16bf b2 = load_bfrag(W0 + (long)(2 * 16) * K + k0, K, lane);
    v16bf b3 = load_bfrag(W0 + (long)(3 * 16) * K + k0, K, lane);
    acc[0][0] = wmma_bf16(a0, b0, acc[0][0]);
    acc[1][0] = wmma_bf16(a1, b0, acc[1][0]);
    acc[0][1] = wmma_bf16(a0, b1, acc[0][1]);
    acc[1][1] = wmma_bf16(a1, b1, acc[1][1]);
    acc[0][2] = wmma_bf16(a0, b2, acc[0][2]);
    acc[1][2] = wmma_bf16(a1, b2, acc[1][2]);
    acc[0][3] = wmma_bf16(a0, b3, acc[0][3]);
    acc[1][3] = wmma_bf16(a1, b3, acc[1][3]);
  }

  int col    = lane & 15;
  int rowoff = (lane & 16) ? 8 : 0;
#pragma unroll
  for (int mi = 0; mi < 2; ++mi) {
#pragma unroll
    for (int j = 0; j < 4; ++j) {
#pragma unroll
      for (int v = 0; v < 8; ++v) {
        int r = mt * 32 + mi * 16 + rowoff + v;
        C[(long)r * N + nt * 64 + j * 16 + col] = acc[mi][j][v];
      }
    }
  }
}

// ---------------------------------------------------------------- QKV epilogue
// RMS-norm + RoPE for q,k -> qh/kh [b][h][t][d] bf16; v mix -> vt [b][h][d][t] bf16 (transposed).
__global__ void qkv_epilogue(const float* __restrict__ qkv, const float* __restrict__ ve,
                             const float* __restrict__ lambdas,
                             bf16_t* __restrict__ qh, bf16_t* __restrict__ kh,
                             bf16_t* __restrict__ vt) {
  __shared__ float sbuf[64];
  int j  = threadIdx.x;                 // 0..63: handles d=j and d=j+64
  long blk = blockIdx.x;                // ((b*T + t)*H + h)
  int h = (int)(blk % H_);
  long bt = blk / H_;
  int t = (int)(bt % T_);
  int b = (int)(bt / T_);

  const float* row = qkv + bt * (3 * HDIM_);
  float q1 = row[h * D_ + j],             q2 = row[h * D_ + j + 64];
  float k1 = row[HDIM_ + h * D_ + j],     k2 = row[HDIM_ + h * D_ + j + 64];
  float v1 = row[2 * HDIM_ + h * D_ + j], v2 = row[2 * HDIM_ + h * D_ + j + 64];

  const float eps = 1.1920929e-7f;
  sbuf[j] = q1 * q1 + q2 * q2;
  __syncthreads();
  for (int s = 32; s > 0; s >>= 1) { if (j < s) sbuf[j] += sbuf[j + s]; __syncthreads(); }
  float rq = rsqrtf(sbuf[0] * (1.0f / 128.0f) + eps);
  __syncthreads();
  sbuf[j] = k1 * k1 + k2 * k2;
  __syncthreads();
  for (int s = 32; s > 0; s >>= 1) { if (j < s) sbuf[j] += sbuf[j + s]; __syncthreads(); }
  float rk = rsqrtf(sbuf[0] * (1.0f / 128.0f) + eps);

  // RoPE tables: freq[j] = 1024^(-j/31) for j<32, else 0 -> cos=1, sin=0.
  float c = 1.0f, s = 0.0f;
  if (j < 32) {
    float th = (float)t * __powf(1024.0f, -(float)j / 31.0f);
    c = __cosf(th); s = __sinf(th);
  }
  float qa = q1 * rq, qb = q2 * rq;
  float ka = k1 * rk, kb = k2 * rk;
  float qo1 =  qa * c + qb * s, qo2 = -qa * s + qb * c;
  float ko1 =  ka * c + kb * s, ko2 = -ka * s + kb * c;

  long base = ((long)(b * H_ + h) * T_ + t) * D_;
  qh[base + j]      = (bf16_t)qo1;
  qh[base + j + 64] = (bf16_t)qo2;
  kh[base + j]      = (bf16_t)ko1;
  kh[base + j + 64] = (bf16_t)ko2;

  float l0 = lambdas[0], l1 = lambdas[1];
  const float* vrow = ve + bt * HDIM_ + h * D_;
  float vo1 = l0 * v1 + l1 * vrow[j];
  float vo2 = l0 * v2 + l1 * vrow[j + 64];
  long vtb = (long)(b * H_ + h) * D_ * T_;
  vt[vtb + (long)j * T_ + t]        = (bf16_t)vo1;
  vt[vtb + (long)(j + 64) * T_ + t] = (bf16_t)vo2;
}

// ---------------------------------------------------------------- fused flash attention
// One wave per 16-row query tile; key blocks of 32; online softmax; all matmuls via WMMA.
__global__ void flash_attn(const bf16_t* __restrict__ qh, const bf16_t* __restrict__ kh,
                           const bf16_t* __restrict__ vt, bf16_t* __restrict__ yh) {
  __shared__ __align__(64) bf16_t plds[4][16 * 32];   // per-wave P tile (16x32 bf16)
  int lane = threadIdx.x & 31;
  int wave = threadIdx.x >> 5;
  long blk = blockIdx.x;                 // (b*H + h) * (T/64) + qgroup
  int qg = (int)(blk % (T_ / 64));
  int bh = (int)(blk / (T_ / 64));
  int q0 = qg * 64 + wave * 16;

  const bf16_t* Q  = qh + (long)bh * T_ * D_;
  const bf16_t* Kp = kh + (long)bh * T_ * D_;
  const bf16_t* Vt = vt + (long)bh * D_ * T_;

  v16bf qf[4];
#pragma unroll
  for (int c = 0; c < 4; ++c)
    qf[c] = load_afrag(Q + (long)q0 * D_ + c * 32, D_, lane);

  v8f o[8];
#pragma unroll
  for (int n = 0; n < 8; ++n) o[n] = (v8f){};
  float mrow[8], lrow[8];
#pragma unroll
  for (int v = 0; v < 8; ++v) { mrow[v] = -1e30f; lrow[v] = 0.0f; }

  const float scale = 0.08838834764831845f;   // 1/sqrt(128)
  int col    = lane & 15;
  int rowoff = (lane & 16) ? 8 : 0;
  bf16_t* pl = plds[wave];

  for (int kb0 = 0; kb0 <= q0 + 15; kb0 += 32) {
    // S = Q * K^T for 32 keys (two 16-col WMMA tiles, K-dim = 128 in 4 chunks)
    v8f s0 = (v8f){}, s1 = (v8f){};
#pragma unroll
    for (int c = 0; c < 4; ++c) {
      v16bf b0 = load_bfrag(Kp + (long)kb0 * D_ + c * 32, D_, lane);
      s0 = wmma_bf16(qf[c], b0, s0);
      v16bf b1 = load_bfrag(Kp + (long)(kb0 + 16) * D_ + c * 32, D_, lane);
      s1 = wmma_bf16(qf[c], b1, s1);
    }
    // scale + causal mask
#pragma unroll
    for (int v = 0; v < 8; ++v) {
      int r = q0 + rowoff + v;
      float a0 = s0[v] * scale; if (kb0 + col > r)      a0 = -1e30f;
      float a1 = s1[v] * scale; if (kb0 + 16 + col > r) a1 = -1e30f;
      s0[v] = a0; s1[v] = a1;
    }
    // online softmax: row lives in one 16-lane group per VGPR
    float alpha[8];
#pragma unroll
    for (int v = 0; v < 8; ++v) {
      float rm = fmaxf(s0[v], s1[v]);
#pragma unroll
      for (int off = 8; off > 0; off >>= 1) rm = fmaxf(rm, __shfl_xor(rm, off, 16));
      float mn = fmaxf(mrow[v], rm);
      alpha[v] = __expf(mrow[v] - mn);
      mrow[v] = mn;
      float p0 = __expf(s0[v] - mn), p1 = __expf(s1[v] - mn);
      s0[v] = p0; s1[v] = p1;
      float rs = p0 + p1;
#pragma unroll
      for (int off = 8; off > 0; off >>= 1) rs += __shfl_xor(rs, off, 16);
      lrow[v] = lrow[v] * alpha[v] + rs;
    }
    // C-layout -> A-layout transpose of P through LDS (same-wave DS ops are in-order)
#pragma unroll
    for (int v = 0; v < 8; ++v) {
      pl[(rowoff + v) * 32 + col]      = (bf16_t)s0[v];
      pl[(rowoff + v) * 32 + 16 + col] = (bf16_t)s1[v];
    }
    __asm volatile("" ::: "memory");
    __builtin_amdgcn_wave_barrier();
    v16bf pf = load_afrag(pl, 32, lane);
    __asm volatile("" ::: "memory");
    __builtin_amdgcn_wave_barrier();
    // rescale accumulators and O += P * V  (V transposed: contiguous B-frag loads)
#pragma unroll
    for (int n = 0; n < 8; ++n) {
#pragma unroll
      for (int v = 0; v < 8; ++v) o[n][v] *= alpha[v];
      v16bf bv = load_bfrag(Vt + (long)(n * 16) * T_ + kb0, T_, lane);
      o[n] = wmma_bf16(pf, bv, o[n]);
    }
  }

  int b = bh / H_, h = bh % H_;
#pragma unroll
  for (int n = 0; n < 8; ++n) {
#pragma unroll
    for (int v = 0; v < 8; ++v) {
      int t = q0 + rowoff + v;
      float val = o[n][v] / lrow[v];
      yh[((long)(b * T_ + t)) * HDIM_ + h * D_ + n * 16 + col] = (bf16_t)val;
    }
  }
}

// ---------------------------------------------------------------- launch
extern "C" void kernel_launch(void* const* d_in, const int* in_sizes, int n_in,
                              void* d_out, int out_size, void* d_ws, size_t ws_size,
                              hipStream_t stream) {
  const float* x       = (const float*)d_in[0];   // [B,T,DIM]
  const float* ve      = (const float*)d_in[1];   // [B,T,HDIM]
  const float* qkv_w   = (const float*)d_in[2];   // [3,HDIM,DIM]
  const float* lambdas = (const float*)d_in[3];   // [2]
  const float* cproj_w = (const float*)d_in[4];   // [DIM,HDIM]
  float* out = (float*)d_out;                     // [B,T,DIM] f32

  const long nx  = (long)B_ * T_ * DIM_;          // 8388608
  const long nwq = 3L * HDIM_ * DIM_;             // 3145728
  const long nwp = (long)DIM_ * HDIM_;            // 1048576
  const long nqkv = (long)B_ * T_ * 3 * HDIM_;    // 25165824
  const long nhd = (long)B_ * H_ * T_ * D_;       // 8388608

  char* w = (char*)d_ws;
  bf16_t* xb    = (bf16_t*)w;                 w += nx  * 2;
  bf16_t* wqkv  = (bf16_t*)w;                 w += nwq * 2;
  bf16_t* wproj = (bf16_t*)w;                 w += nwp * 2;
  float*  qkv   = (float*)w;                  w += nqkv * 4;
  bf16_t* qh    = (bf16_t*)w;                 w += nhd * 2;
  bf16_t* kh    = (bf16_t*)w;                 w += nhd * 2;
  bf16_t* vt    = (bf16_t*)w;                 w += nhd * 2;
  bf16_t* yh    = (bf16_t*)w;                 w += nhd * 2;   // ~184 MB total

  // 1) fp32 -> bf16
  cvt_f32_to_bf16<<<(int)((nx  + 255) / 256), 256, 0, stream>>>(x, xb, nx);
  cvt_f32_to_bf16<<<(int)((nwq + 255) / 256), 256, 0, stream>>>(qkv_w, wqkv, nwq);
  cvt_f32_to_bf16<<<(int)((nwp + 255) / 256), 256, 0, stream>>>(cproj_w, wproj, nwp);

  // 2) QKV GEMM: [8192 x 1024] * [3072 x 1024]^T -> [8192 x 3072] f32
  {
    int M = B_ * T_, N = 3 * HDIM_, K = DIM_;
    long tiles = (long)(M / 32) * (N / 64);     // 32x64 macro-tile per wave
    gemm_bf16_abt<<<(int)((tiles + 3) / 4), 128, 0, stream>>>(xb, wqkv, qkv, M, N, K);
  }

  // 3) epilogue: RMS-norm + RoPE (q,k), lambda mix + transpose (v)
  qkv_epilogue<<<B_ * T_ * H_, 64, 0, stream>>>(qkv, ve, lambdas, qh, kh, vt);

  // 4) fused flash attention: B*H*(T/64) blocks, 4 waves (q-tiles) each
  flash_attn<<<B_ * H_ * (T_ / 64), 128, 0, stream>>>(qh, kh, vt, yh);

  // 5) output projection: [8192 x 1024] * [1024 x 1024]^T -> out f32
  {
    int M = B_ * T_, N = DIM_, K = HDIM_;
    long tiles = (long)(M / 32) * (N / 64);
    gemm_bf16_abt<<<(int)((tiles + 3) / 4), 128, 0, stream>>>(yh, wproj, out, M, N, K);
  }
}
